// RelationalNetwork_78563541778630
// MI455X (gfx1250) — matrix operand order repaired and checked by reading, not compiled, not run on hardware
//
#include <hip/hip_runtime.h>

// ---------------- types ----------------
typedef __attribute__((ext_vector_type(16))) _Float16 v16h;
typedef __attribute__((ext_vector_type(8)))  float    v8f;

union F16Frag { uint4 u[2]; v16h v; };

#define MB   64
#define QLEN 20
#define EMBD 32
#define QVS  128
#define ANSN 10
#define CHN  24
#define NOBJ 64
#define OBJF 26
#define GD   256

__device__ __forceinline__ float sigm_(float x) { return 1.0f / (1.0f + expf(-x)); }

// Add an opaque zero byte-offset to a pointer. The base pointer keeps its
// address-space provenance (loads stay global_load_*, not flat_*), but the
// address depends on an asm-defined value, so LICM cannot hoist loads out of
// the enclosing loop (which previously caused register spills to scratch).
template <typename T>
__device__ __forceinline__ const T* launder_ptr(const T* p) {
    unsigned long long off = 0;
    asm volatile("" : "+s"(off));
    return (const T*)((const char*)p + off);
}

// ---------------- init: zero accumulators ----------------
__global__ void init_zero_kernel(float* stats, float* x_g) {
    int i = blockIdx.x * 256 + threadIdx.x;
    if (i < 256)   stats[i] = 0.0f;
    if (i < MB*GD) x_g[i]   = 0.0f;
}

// ---------------- pack g2/g3/g4 weights to f16 ----------------
__global__ void pack_gw_kernel(const float* __restrict__ g2, const float* __restrict__ g3,
                               const float* __restrict__ g4,
                               _Float16* w2, _Float16* w3, _Float16* w4) {
    int i = blockIdx.x * 256 + threadIdx.x;
    if (i < GD*GD) {
        w2[i] = (_Float16)g2[i];
        w3[i] = (_Float16)g3[i];
        w4[i] = (_Float16)g4[i];
    }
}

// ---------------- pack LSTM weights: wcomb[u][k], k<32 = wih, else whh ----------------
__global__ void pack_lstm_kernel(const float* __restrict__ wih, const float* __restrict__ whh,
                                 _Float16* wcomb) {
    int i = blockIdx.x * 256 + threadIdx.x;
    if (i < 4*QVS*(EMBD+QVS)) {
        int k = i % (EMBD+QVS);
        int u = i / (EMBD+QVS);
        float v = (k < EMBD) ? wih[u*EMBD + k] : whh[u*QVS + (k-EMBD)];
        wcomb[i] = (_Float16)v;
    }
}

// ---------------- embedding lookup -> f16, layout (t, b, k) ----------------
__global__ void embed_kernel(const int* __restrict__ question, const float* __restrict__ emb,
                             _Float16* xemb) {
    int i = blockIdx.x * 256 + threadIdx.x;
    if (i < QLEN*MB*EMBD) {
        int k = i & (EMBD-1);
        int b = (i >> 5) & (MB-1);
        int t = i >> 11;
        int tok = question[b*QLEN + t];
        xemb[i] = (_Float16)emb[tok*EMBD + k];
    }
}

// ---------------- LSTM: single WG, WMMA per step (64x512 @ K=160) ----------------
__global__ __launch_bounds__(256) void lstm_kernel(const _Float16* __restrict__ xemb,
                                                   const _Float16* __restrict__ wcomb,
                                                   const float* __restrict__ bih,
                                                   const float* __restrict__ bhh,
                                                   float* __restrict__ qvec) {
    extern __shared__ char smem[];
    const int AS = 168;                              // padded row stride (halves), 336B (16B-aligned)
    _Float16* A  = (_Float16*)smem;                  // [64][168] : cols 0..31 = x_t, 32..159 = h
    float* gates = (float*)(smem + 64*AS*2);         // [64][512]
    float* cst   = (float*)(smem + 64*AS*2 + 64*512*4); // [64][128]

    int tid = threadIdx.x;
    for (int idx = tid; idx < MB*QVS; idx += 256) {
        int b = idx >> 7, u = idx & 127;
        A[b*AS + EMBD + u] = (_Float16)0.0f;
        cst[idx] = 0.0f;
    }
    __syncthreads();

    int w = tid >> 5, lane = tid & 31;
    int hi = lane >> 4, lm = lane & 15;
    int m0 = (w & 3) * 16;
    int grp = w >> 2;                                // 0/1 -> Ntiles [0..15] / [16..31]

    for (int t = 0; t < QLEN; t++) {
        // Opaque per-iteration bases: weight/bias loads stay inside the step
        // loop (served from L2 via global_load) instead of hoisted+spilled.
        const _Float16* wct  = launder_ptr(wcomb);
        const float*    bihL = launder_ptr(bih);
        const float*    bhhL = launder_ptr(bhh);

        for (int idx = tid; idx < MB*EMBD; idx += 256) {
            int b = idx >> 5, k = idx & 31;
            A[b*AS + k] = xemb[(t*MB + b)*EMBD + k];
        }
        __syncthreads();

        #pragma unroll 1
        for (int nti = 0; nti < 16; nti++) {
            int nt = grp*16 + nti;
            v8f acc = {};
            #pragma unroll
            for (int kt = 0; kt < 5; kt++) {
                F16Frag a, bf;
                const _Float16* ap = A + (m0 + lm)*AS + kt*32 + hi*8;
                a.u[0] = *(const uint4*)(ap);
                a.u[1] = *(const uint4*)(ap + 16);
                const _Float16* wp = wct + (nt*16 + lm)*(EMBD+QVS) + kt*32 + hi*8;
                bf.u[0] = *(const uint4*)(wp);
                bf.u[1] = *(const uint4*)(wp + 16);
                acc = __builtin_amdgcn_wmma_f32_16x16x32_f16(
                        false, a.v, false, bf.v, (short)0, acc, false, false);
            }
            int n = nt*16 + lm;
            #pragma unroll
            for (int r = 0; r < 8; r++)
                gates[(m0 + r + 8*hi)*512 + n] = acc[r];
        }
        __syncthreads();

        for (int idx = tid; idx < MB*QVS; idx += 256) {
            int b = idx >> 7, u = idx & 127;
            float ig = gates[b*512 +       u] + bihL[      u] + bhhL[      u];
            float fg = gates[b*512 + 128 + u] + bihL[128 + u] + bhhL[128 + u];
            float gg = gates[b*512 + 256 + u] + bihL[256 + u] + bhhL[256 + u];
            float og = gates[b*512 + 384 + u] + bihL[384 + u] + bhhL[384 + u];
            float c  = sigm_(fg)*cst[idx] + sigm_(ig)*tanhf(gg);
            float h  = sigm_(og)*tanhf(c);
            cst[idx] = c;
            A[b*AS + EMBD + u] = (_Float16)h;
            if (t == QLEN-1) qvec[b*QVS + u] = h;
        }
        __syncthreads();
    }
}

// ---------------- conv 3x3 stride2 pad1 + batch-stat accumulation ----------------
__global__ __launch_bounds__(256) void conv_kernel(const float* __restrict__ x,
                                                   const float* __restrict__ wt,
                                                   const float* __restrict__ bias,
                                                   float* __restrict__ y,
                                                   float* __restrict__ stats,
                                                   int N, int Cin, int Hin, int Win,
                                                   int Cout, int Hout, int Wout) {
    __shared__ float ssum[32], ssq[32];
    if (threadIdx.x < 32) { ssum[threadIdx.x] = 0.0f; ssq[threadIdx.x] = 0.0f; }
    __syncthreads();
    int total = N*Cout*Hout*Wout;
    int idx = blockIdx.x * 256 + threadIdx.x;
    if (idx < total) {
        int wo = idx % Wout; int t = idx / Wout;
        int ho = t % Hout;  t /= Hout;
        int co = t % Cout;  int n = t / Cout;
        float acc = bias[co];
        int hb = ho*2 - 1, wb = wo*2 - 1;
        for (int ci = 0; ci < Cin; ci++) {
            const float* xp = x  + ((size_t)(n*Cin + ci))*Hin*Win;
            const float* wp = wt + ((size_t)(co*Cin + ci))*9;
            #pragma unroll
            for (int kh = 0; kh < 3; kh++) {
                int hy = hb + kh;
                if (hy < 0 || hy >= Hin) continue;
                #pragma unroll
                for (int kw = 0; kw < 3; kw++) {
                    int wx = wb + kw;
                    if (wx < 0 || wx >= Win) continue;
                    acc += xp[hy*Win + wx] * wp[kh*3 + kw];
                }
            }
        }
        y[idx] = acc;
        atomicAdd(&ssum[co], acc);
        atomicAdd(&ssq[co], acc*acc);
    }
    __syncthreads();
    if (threadIdx.x < Cout) {
        atomicAdd(&stats[threadIdx.x],      ssum[threadIdx.x]);
        atomicAdd(&stats[32 + threadIdx.x], ssq[threadIdx.x]);
    }
}

// ---------------- BN (batch stats) + ReLU, in place ----------------
__global__ __launch_bounds__(256) void bn_relu_kernel(float* __restrict__ y,
                                                      const float* __restrict__ stats,
                                                      const float* __restrict__ gamma,
                                                      const float* __restrict__ beta,
                                                      int C, int HW, float invCnt, int total) {
    int idx = blockIdx.x * 256 + threadIdx.x;
    if (idx < total) {
        int c = (idx / HW) % C;
        float mean = stats[c] * invCnt;
        float var  = stats[32 + c] * invCnt - mean*mean;
        float v = (y[idx] - mean) * rsqrtf(var + 1e-5f) * gamma[c] + beta[c];
        y[idx] = fmaxf(v, 0.0f);
    }
}

// ---------------- pre_q = qvec @ w_q.T + g1_b ----------------
__global__ __launch_bounds__(256) void preq_kernel(const float* __restrict__ qvec,
                                                   const float* __restrict__ g1_w,
                                                   const float* __restrict__ g1_b,
                                                   float* __restrict__ pre_q) {
    __shared__ float qr[QVS];
    int b = blockIdx.x, tid = threadIdx.x;
    if (tid < QVS) qr[tid] = qvec[b*QVS + tid];
    __syncthreads();
    const float* gw = g1_w + tid*(2*OBJF + QVS) + 2*OBJF;
    float s = g1_b[tid];
    for (int k = 0; k < QVS; k++) s += qr[k] * gw[k];
    pre_q[b*GD + tid] = s;
}

// ---------------- pre_i / pre_j per object ----------------
__global__ __launch_bounds__(256) void pre_obj_kernel(const float* __restrict__ x4,
                                                      const float* __restrict__ g1_w,
                                                      float* __restrict__ pre_i,
                                                      float* __restrict__ pre_j) {
    __shared__ float obj[OBJF];
    int bo = blockIdx.x;
    int b = bo >> 6, o = bo & 63;
    int tid = threadIdx.x;
    if (tid < OBJF) {
        float v;
        if (tid < CHN)       v = x4[((size_t)(b*CHN + tid)*8 + (o >> 3))*8 + (o & 7)];
        else if (tid == 24)  v = ((o >> 3) - 2) * 0.5f;
        else                 v = ((o & 7) - 2) * 0.5f;
        obj[tid] = v;
    }
    __syncthreads();
    const float* gw = g1_w + tid*(2*OBJF + QVS);
    float si = 0.0f, sj = 0.0f;
    for (int c = 0; c < OBJF; c++) {
        si += obj[c] * gw[c];
        sj += obj[c] * gw[OBJF + c];
    }
    pre_i[(size_t)bo*GD + tid] = si;
    pre_j[(size_t)bo*GD + tid] = sj;
}

// ---------------- main RN kernel: per (b,j), fused 3x [64x256 @ 256x256] via WMMA ----------------
__global__ __launch_bounds__(256) void rn_main_kernel(const float* __restrict__ pre_i,
                                                      const float* __restrict__ pre_j,
                                                      const float* __restrict__ pre_q,
                                                      const _Float16* __restrict__ w2,
                                                      const _Float16* __restrict__ w3,
                                                      const _Float16* __restrict__ w4,
                                                      const float* __restrict__ b2,
                                                      const float* __restrict__ b3,
                                                      const float* __restrict__ b4,
                                                      float* __restrict__ x_g) {
    const int AS = 264;                               // padded row stride (halves), 528B
    __shared__ _Float16 Abuf[64*AS];
    __shared__ float base[GD];
    __shared__ float sums[GD];

    int tid = threadIdx.x;
    int b = blockIdx.x >> 6, j = blockIdx.x & 63;

    base[tid] = pre_j[((size_t)(b*NOBJ + j))*GD + tid] + pre_q[b*GD + tid];
    sums[tid] = 0.0f;
    __syncthreads();

    for (int idx = tid; idx < NOBJ*GD; idx += 256) {
        int i = idx >> 8, c = idx & 255;
        float v = pre_i[((size_t)(b*NOBJ + i))*GD + c] + base[c];
        Abuf[i*AS + c] = (_Float16)fmaxf(v, 0.0f);
    }
    __syncthreads();

    int w = tid >> 5, lane = tid & 31;
    int hi = lane >> 4, lm = lane & 15;
    int m0 = (w & 3) * 16;
    int nh = w >> 2;                                  // 0/1 -> Ntiles [0..7]/[8..15]

    const _Float16* Wts[3] = { w2, w3, w4 };
    const float*    Bs[3]  = { b2, b3, b4 };

    for (int L = 0; L < 3; L++) {
        const _Float16* W    = Wts[L];
        const float*    bias = Bs[L];

        // Warm L2/WGP$ with the next layer's weights while this layer computes.
        if (L < 2) __builtin_prefetch(Wts[L+1] + (size_t)tid*256, 0, 3);

        v8f acc[8];
        #pragma unroll
        for (int q = 0; q < 8; q++) { v8f z = {}; acc[q] = z; }

        for (int kt = 0; kt < 8; kt++) {
            F16Frag a;
            const _Float16* ap = Abuf + (m0 + lm)*AS + kt*32 + hi*8;
            a.u[0] = *(const uint4*)(ap);
            a.u[1] = *(const uint4*)(ap + 16);
            #pragma unroll
            for (int q = 0; q < 8; q++) {
                int n = (nh*8 + q)*16 + lm;
                F16Frag bf;
                const _Float16* wp = W + n*GD + kt*32 + hi*8;
                bf.u[0] = *(const uint4*)(wp);
                bf.u[1] = *(const uint4*)(wp + 16);
                acc[q] = __builtin_amdgcn_wmma_f32_16x16x32_f16(
                           false, a.v, false, bf.v, (short)0, acc[q], false, false);
            }
        }
        __syncthreads();   // all waves done reading Abuf

        #pragma unroll
        for (int q = 0; q < 8; q++) {
            int n = (nh*8 + q)*16 + lm;
            float bv = bias[n];
            if (L < 2) {
                #pragma unroll
                for (int r = 0; r < 8; r++) {
                    float v = fmaxf(acc[q][r] + bv, 0.0f);
                    Abuf[(m0 + r + 8*hi)*AS + n] = (_Float16)v;
                }
            } else {
                float s = 0.0f;
                #pragma unroll
                for (int r = 0; r < 8; r++) s += fmaxf(acc[q][r] + bv, 0.0f);
                atomicAdd(&sums[n], s);
            }
        }
        __syncthreads();
    }

    atomicAdd(&x_g[b*GD + tid], sums[tid]);
}

// ---------------- final MLP + log_softmax ----------------
__global__ __launch_bounds__(256) void final_kernel(const float* __restrict__ x_g,
                                                    const float* __restrict__ f1_w, const float* __restrict__ f1_b,
                                                    const float* __restrict__ f2_w, const float* __restrict__ f2_b,
                                                    const float* __restrict__ f3_w, const float* __restrict__ f3_b,
                                                    float* __restrict__ out) {
    __shared__ float xr[GD], h1[GD], h2[GD], logits[ANSN];
    __shared__ float lse;
    int b = blockIdx.x, tid = threadIdx.x;

    xr[tid] = x_g[b*GD + tid];
    __syncthreads();
    {
        const float* wp = f1_w + tid*GD;
        float s = f1_b[tid];
        for (int k = 0; k < GD; k++) s += xr[k] * wp[k];
        h1[tid] = fmaxf(s, 0.0f);
    }
    __syncthreads();
    {
        const float* wp = f2_w + tid*GD;
        float s = f2_b[tid];
        for (int k = 0; k < GD; k++) s += h1[k] * wp[k];
        h2[tid] = fmaxf(s, 0.0f);
    }
    __syncthreads();
    if (tid < ANSN) {
        const float* wp = f3_w + tid*GD;
        float s = f3_b[tid];
        for (int k = 0; k < GD; k++) s += h2[k] * wp[k];
        logits[tid] = s;
    }
    __syncthreads();
    if (tid == 0) {
        float m = logits[0];
        for (int a = 1; a < ANSN; a++) m = fmaxf(m, logits[a]);
        float se = 0.0f;
        for (int a = 0; a < ANSN; a++) se += expf(logits[a] - m);
        lse = m + logf(se);
    }
    __syncthreads();
    if (tid < ANSN) out[b*ANSN + tid] = logits[tid] - lse;
}

// ---------------- host launcher ----------------
extern "C" void kernel_launch(void* const* d_in, const int* in_sizes, int n_in,
                              void* d_out, int out_size, void* d_ws, size_t ws_size,
                              hipStream_t stream) {
    const float* image    = (const float*)d_in[0];
    const int*   question = (const int*)  d_in[1];
    const float* emb      = (const float*)d_in[2];
    const float* lstm_wih = (const float*)d_in[3];
    const float* lstm_whh = (const float*)d_in[4];
    const float* lstm_bih = (const float*)d_in[5];
    const float* lstm_bhh = (const float*)d_in[6];
    const float* cw[4] = { (const float*)d_in[7],  (const float*)d_in[11], (const float*)d_in[15], (const float*)d_in[19] };
    const float* cb[4] = { (const float*)d_in[8],  (const float*)d_in[12], (const float*)d_in[16], (const float*)d_in[20] };
    const float* bg[4] = { (const float*)d_in[9],  (const float*)d_in[13], (const float*)d_in[17], (const float*)d_in[21] };
    const float* bb[4] = { (const float*)d_in[10], (const float*)d_in[14], (const float*)d_in[18], (const float*)d_in[22] };
    const float* g1_w = (const float*)d_in[23];
    const float* g1_b = (const float*)d_in[24];
    const float* g2_w = (const float*)d_in[25];
    const float* g2_b = (const float*)d_in[26];
    const float* g3_w = (const float*)d_in[27];
    const float* g3_b = (const float*)d_in[28];
    const float* g4_w = (const float*)d_in[29];
    const float* g4_b = (const float*)d_in[30];
    const float* f1_w = (const float*)d_in[31];
    const float* f1_b = (const float*)d_in[32];
    const float* f2_w = (const float*)d_in[33];
    const float* f2_b = (const float*)d_in[34];
    const float* f3_w = (const float*)d_in[35];
    const float* f3_b = (const float*)d_in[36];

    // ---- workspace carving ----
    char* p = (char*)d_ws;
    auto carve = [&](size_t bytes) -> void* {
        void* r = (void*)p;
        p += (bytes + 255) & ~(size_t)255;
        return r;
    };
    float* x1    = (float*)carve((size_t)64*24*64*64*4);   // 25.2 MB
    float* x2    = (float*)carve((size_t)64*24*32*32*4);   //  6.3 MB
    float* x3    = (float*)carve((size_t)64*24*16*16*4);
    float* x4    = (float*)carve((size_t)64*24*8*8*4);
    float* stats = (float*)carve(256*4);                   // 4 layers x {sum[32], sumsq[32]}
    float* qvec  = (float*)carve((size_t)MB*QVS*4);
    float* pre_q = (float*)carve((size_t)MB*GD*4);
    float* pre_i = (float*)carve((size_t)MB*NOBJ*GD*4);    // 4 MB
    float* pre_j = (float*)carve((size_t)MB*NOBJ*GD*4);    // 4 MB
    float* x_g   = (float*)carve((size_t)MB*GD*4);
    _Float16* w2h   = (_Float16*)carve((size_t)GD*GD*2);
    _Float16* w3h   = (_Float16*)carve((size_t)GD*GD*2);
    _Float16* w4h   = (_Float16*)carve((size_t)GD*GD*2);
    _Float16* wcomb = (_Float16*)carve((size_t)4*QVS*(EMBD+QVS)*2);
    _Float16* xemb  = (_Float16*)carve((size_t)QLEN*MB*EMBD*2);

    // ---- prep ----
    init_zero_kernel<<<(MB*GD + 255)/256, 256, 0, stream>>>(stats, x_g);
    pack_gw_kernel<<<(GD*GD + 255)/256, 256, 0, stream>>>(g2_w, g3_w, g4_w, w2h, w3h, w4h);
    pack_lstm_kernel<<<(4*QVS*(EMBD+QVS) + 255)/256, 256, 0, stream>>>(lstm_wih, lstm_whh, wcomb);
    embed_kernel<<<(QLEN*MB*EMBD + 255)/256, 256, 0, stream>>>(question, emb, xemb);

    // ---- LSTM (single WG, dynamic LDS = A + gates + c state) ----
    size_t lstm_lds = (size_t)64*168*2 + (size_t)64*512*4 + (size_t)64*128*4;
    lstm_kernel<<<1, 256, lstm_lds, stream>>>(xemb, wcomb, lstm_bih, lstm_bhh, qvec);

    // ---- conv stack ----
    {   // layer 1: (64,3,128,128) -> (64,24,64,64)
        int total = 64*24*64*64;
        conv_kernel<<<(total + 255)/256, 256, 0, stream>>>(image, cw[0], cb[0], x1, stats + 0,
                                                           64, 3, 128, 128, 24, 64, 64);
        bn_relu_kernel<<<(total + 255)/256, 256, 0, stream>>>(x1, stats + 0, bg[0], bb[0],
                                                              24, 64*64, 1.0f/(64.0f*64*64), total);
    }
    {   // layer 2 -> (64,24,32,32)
        int total = 64*24*32*32;
        conv_kernel<<<(total + 255)/256, 256, 0, stream>>>(x1, cw[1], cb[1], x2, stats + 64,
                                                           64, 24, 64, 64, 24, 32, 32);
        bn_relu_kernel<<<(total + 255)/256, 256, 0, stream>>>(x2, stats + 64, bg[1], bb[1],
                                                              24, 32*32, 1.0f/(64.0f*32*32), total);
    }
    {   // layer 3 -> (64,24,16,16)
        int total = 64*24*16*16;
        conv_kernel<<<(total + 255)/256, 256, 0, stream>>>(x2, cw[2], cb[2], x3, stats + 128,
                                                           64, 24, 32, 32, 24, 16, 16);
        bn_relu_kernel<<<(total + 255)/256, 256, 0, stream>>>(x3, stats + 128, bg[2], bb[2],
                                                              24, 16*16, 1.0f/(64.0f*16*16), total);
    }
    {   // layer 4 -> (64,24,8,8)
        int total = 64*24*8*8;
        conv_kernel<<<(total + 255)/256, 256, 0, stream>>>(x3, cw[3], cb[3], x4, stats + 192,
                                                           64, 24, 16, 16, 24, 8, 8);
        bn_relu_kernel<<<(total + 255)/256, 256, 0, stream>>>(x4, stats + 192, bg[3], bb[3],
                                                              24, 8*8, 1.0f/(64.0f*8*8), total);
    }

    // ---- RN front end ----
    preq_kernel<<<MB, 256, 0, stream>>>(qvec, g1_w, g1_b, pre_q);
    pre_obj_kernel<<<MB*NOBJ, 256, 0, stream>>>(x4, g1_w, pre_i, pre_j);

    // ---- fused g2/g3/g4 WMMA kernel: one WG per (b, j) ----
    rn_main_kernel<<<MB*NOBJ, 256, 0, stream>>>(pre_i, pre_j, pre_q, w2h, w3h, w4h,
                                                g2_b, g3_b, g4_b, x_g);

    // ---- final MLP + log_softmax ----
    final_kernel<<<MB, 256, 0, stream>>>(x_g, f1_w, f1_b, f2_w, f2_b, f3_w, f3_b, (float*)d_out);
}
